// InstantNGP_19138374271629
// MI455X (gfx1250) — matrix operand order repaired and checked by reading, not compiled, not run on hardware
//
#include <hip/hip_runtime.h>
#include <hip/hip_bf16.h>

typedef __attribute__((ext_vector_type(16))) _Float16 v16h;
typedef __attribute__((ext_vector_type(8)))  float    v8f;
typedef __attribute__((ext_vector_type(2)))  int      v2i;

typedef __attribute__((address_space(1))) v2i as1_v2i;
typedef __attribute__((address_space(3))) v2i as3_v2i;

// Same-wave LDS ops are pipeline-ordered; this is a compiler barrier + explicit
// dscnt drain so cross-lane LDS RAW between pipeline stages is safe.
#define LDS_FENCE() asm volatile("s_wait_dscnt 0x0" ::: "memory")

#if __has_builtin(__builtin_amdgcn_global_load_async_to_lds_b64) && \
    __has_builtin(__builtin_amdgcn_s_wait_asynccnt)
#define HAVE_ASYNC 1
#define ASYNC_GATHER_B64(g, l)                                             \
  __builtin_amdgcn_global_load_async_to_lds_b64(                           \
      (as1_v2i*)(void*)(g), (as3_v2i*)(void*)(l), 0, 0)
#define WAIT_ASYNC(n)                         \
  do {                                        \
    __builtin_amdgcn_s_wait_asynccnt(n);      \
    asm volatile("" ::: "memory");            \
  } while (0)
#else
#define HAVE_ASYNC 0
#endif

namespace {
constexpr int      kN     = 262144;
constexpr int      kTiles = kN / 16;
constexpr unsigned kT     = 1u << 19;
constexpr float    kStep  = 0.0016914558f; // sqrt(3)/1024
}

// floor(16 * 1.5^l) for l = 0..15
__device__ __constant__ float g_res[16] = {
    16.f, 24.f, 36.f, 54.f, 81.f, 121.f, 182.f, 273.f,
    410.f, 615.f, 922.f, 1383.f, 2075.f, 3113.f, 4670.f, 7006.f};

__device__ inline v8f wmma_f16(v16h a, v16h b, v8f c) {
  return __builtin_amdgcn_wmma_f32_16x16x32_f16(
      /*neg_a=*/false, a, /*neg_b=*/false, b,
      /*c_mod=*/(short)0, c, /*reuse_a=*/false, /*reuse_b=*/false);
}

// A-fragment (16-bit A 16x32 layout): lanes 0-15 hold M=lane, K pairs
// {0..7,16..23}; lanes 16-31 hold M=lane-16, K pairs {8..15,24..31}.
__device__ inline v16h load_a(const _Float16* base, int strideHalves, int kbase) {
  const int lane = threadIdx.x & 31;
  const int m  = lane & 15;
  const int hi = lane >> 4;
  const _Float16* p = base + m * strideHalves + kbase + hi * 8;
  v16h r;
  ((uint4*)&r)[0] = *(const uint4*)(p);        // K = hi*8 + 0..7
  ((uint4*)&r)[1] = *(const uint4*)(p + 16);   // K = 16 + hi*8 + 0..7
  return r;
}

// B-fragment (16-bit B 32x16 layout): lanes 0-15 hold K={0..7,16..23} (N=lane),
// lanes 16-31 hold K={8..15,24..31} (N=lane-16). Source row-major w[k][n], ncols wide.
__device__ inline v16h load_b(const _Float16* w, int ncols, int kbase, int nbase) {
  const int lane = threadIdx.x & 31;
  const int n  = (lane & 15) + nbase;
  const int hi = lane >> 4;
  v16h r;
#pragma unroll
  for (int j = 0; j < 16; ++j)
    r[j] = w[(kbase + hi * 8 + (j < 8 ? j : j + 8)) * ncols + n];
  return r;
}

// C/D (32-bit 16x16): VGPR r holds M = r + (lane>=16 ? 8 : 0), N = lane&15.
__device__ inline void store_c(_Float16* base, int strideHalves, int nbase,
                               v8f c, bool relu) {
  const int lane = threadIdx.x & 31;
  const int n  = (lane & 15) + nbase;
  const int hi = lane >> 4;
#pragma unroll
  for (int r = 0; r < 8; ++r) {
    float v = c[r];
    if (relu) v = fmaxf(v, 0.f);
    base[(r + hi * 8) * strideHalves + n] = (_Float16)v;
  }
}

#if HAVE_ASYNC
// Issue the 8 corner gathers for one (sample,level) pair as async global->LDS
// b64 transfers into slot[c][lane] (conflict-free: lane*8B within each row).
__device__ inline void issue_batch(const float* table, int lv, float px,
                                   float py, float pz, float2* slot, int lane) {
  const float rs = g_res[lv];
  const unsigned cx = (unsigned)(int)floorf(px * rs);
  const unsigned cy = (unsigned)(int)floorf(py * rs);
  const unsigned cz = (unsigned)(int)floorf(pz * rs);
  const float2* tb = ((const float2*)table) + (size_t)lv * kT;
#pragma unroll
  for (int c = 0; c < 8; ++c) {
    const unsigned b2 = (c >> 2) & 1, b1 = (c >> 1) & 1, b0 = c & 1;
    const unsigned h =
        (cx + b2) * 1u ^ (cy + b1) * 2654435761u ^ (cz + b0) * 805459861u;
    ASYNC_GATHER_B64(tb + (h & (kT - 1u)), slot + c * 32 + lane);
  }
}

// Trilinear-blend the 8 gathered corners (now in LDS) and emit 2 f16 features.
__device__ inline void consume_batch(const float2* slot, int lane, int lv,
                                     float px, float py, float pz,
                                     _Float16* scr, int laneLo) {
  const float rs = g_res[lv];
  const float fx = px * rs, fy = py * rs, fz = pz * rs;
  const float rx = fx - floorf(fx), ry = fy - floorf(fy), rz = fz - floorf(fz);
  float a0 = 0.f, a1 = 0.f;
#pragma unroll
  for (int c = 0; c < 8; ++c) {
    const unsigned b2 = (c >> 2) & 1, b1 = (c >> 1) & 1, b0 = c & 1;
    const float2 f = slot[c * 32 + lane];
    const float w = (b2 ? rx : 1.f - rx) * (b1 ? ry : 1.f - ry) *
                    (b0 ? rz : 1.f - rz);
    a0 += w * f.x;
    a1 += w * f.y;
  }
  scr[laneLo * 32 + lv * 2 + 0] = (_Float16)a0;
  scr[laneLo * 32 + lv * 2 + 1] = (_Float16)a1;
}
#endif

__global__ __launch_bounds__(128, 1)
void InstantNGP_fused_kernel(const float* __restrict__ pos,
                             const float* __restrict__ dir,
                             const float* __restrict__ table,
                             const float* __restrict__ w1s,
                             const float* __restrict__ w2s,
                             const float* __restrict__ w1r,
                             const float* __restrict__ w2r,
                             const float* __restrict__ w3r,
                             float* __restrict__ out_rgb,
                             float* __restrict__ out_alpha) {
  __shared__ _Float16 sW1s[32 * 64];
  __shared__ _Float16 sW2s[64 * 16];
  __shared__ _Float16 sW1r[32 * 64];
  __shared__ _Float16 sW2r[64 * 64];
  __shared__ _Float16 sW3r[64 * 16];
  __shared__ _Float16 sScr[4][16 * 64];      // per-wave activation bounce
#if HAVE_ASYNC
  __shared__ float2   sGat[4][2][8][32];     // per-wave dbl-buffered gathers
#endif

  const int tid = threadIdx.x;

  // Stage all weights into LDS as f16 (10240 values, cooperative).
  for (int i = tid; i < 32 * 64; i += 128) sW1s[i] = (_Float16)w1s[i];
  for (int i = tid; i < 64 * 16; i += 128) sW2s[i] = (_Float16)w2s[i];
  for (int i = tid; i < 32 * 64; i += 128) sW1r[i] = (_Float16)w1r[i];
  for (int i = tid; i < 64 * 64; i += 128) sW2r[i] = (_Float16)w2r[i];
  for (int i = tid; i < 64 * 16; i += 128) sW3r[i] = (_Float16)w3r[i];
  __syncthreads();

  // Resident B fragments for all 5 weight matrices (reused for every tile).
  v16h B1s[4], B2s[2], B1r[4], B2r[2][4], B3r[2];
#pragma unroll
  for (int nt = 0; nt < 4; ++nt) B1s[nt] = load_b(sW1s, 64, 0, nt * 16);
#pragma unroll
  for (int kt = 0; kt < 2; ++kt) B2s[kt] = load_b(sW2s, 16, kt * 32, 0);
#pragma unroll
  for (int nt = 0; nt < 4; ++nt) B1r[nt] = load_b(sW1r, 64, 0, nt * 16);
#pragma unroll
  for (int kt = 0; kt < 2; ++kt)
#pragma unroll
    for (int nt = 0; nt < 4; ++nt) B2r[kt][nt] = load_b(sW2r, 64, kt * 32, nt * 16);
#pragma unroll
  for (int kt = 0; kt < 2; ++kt) B3r[kt] = load_b(sW3r, 16, kt * 32, 0);

  const int lane   = tid & 31;
  const int laneLo = lane & 15;
  const int hi     = lane >> 4;
  _Float16* scr = sScr[tid >> 5];

  const int wave  = blockIdx.x * 4 + (tid >> 5);
  const int nwave = gridDim.x * 4;

  for (int tile = wave; tile < kTiles; tile += nwave) {
    const int s0 = tile * 16;

    // ---- Stage 1: hash-grid encode (64MB table resident in 192MB L2) ----
    const float px = pos[(size_t)(s0 + laneLo) * 3 + 0];
    const float py = pos[(size_t)(s0 + laneLo) * 3 + 1];
    const float pz = pos[(size_t)(s0 + laneLo) * 3 + 2];

#if HAVE_ASYNC
    {
      // Double-buffered async gather pipeline: batch k+1 in flight while
      // batch k is blended. Async loads retire in order, so after issuing
      // batch k+1 (8 ops), asynccnt<=8 implies batch k has landed in LDS.
      float2* gbuf = &sGat[tid >> 5][0][0][0];     // [2][8][32]
      issue_batch(table, hi, px, py, pz, gbuf, lane);
      for (int it = 0; it < 7; ++it) {
        issue_batch(table, 2 * (it + 1) + hi, px, py, pz,
                    gbuf + ((it + 1) & 1) * 8 * 32, lane);
        WAIT_ASYNC(8);
        consume_batch(gbuf + (it & 1) * 8 * 32, lane, 2 * it + hi,
                      px, py, pz, scr, laneLo);
      }
      WAIT_ASYNC(0);
      consume_batch(gbuf + (7 & 1) * 8 * 32, lane, 14 + hi,
                    px, py, pz, scr, laneLo);
    }
#else
    for (int it = 0; it < 8; ++it) {
      const int lv = it * 2 + hi;
      const float rs = g_res[lv];
      const float fx = px * rs, fy = py * rs, fz = pz * rs;
      const float bx = floorf(fx), by = floorf(fy), bz = floorf(fz);
      const float rx = fx - bx, ry = fy - by, rz = fz - bz;
      const unsigned cx = (unsigned)(int)bx;
      const unsigned cy = (unsigned)(int)by;
      const unsigned cz = (unsigned)(int)bz;
      const float2* tb = ((const float2*)table) + (size_t)lv * kT;
      float a0 = 0.f, a1 = 0.f;
#pragma unroll
      for (int c = 0; c < 8; ++c) {
        const unsigned b2 = (c >> 2) & 1, b1 = (c >> 1) & 1, b0 = c & 1;
        const unsigned h =
            (cx + b2) * 1u ^ (cy + b1) * 2654435761u ^ (cz + b0) * 805459861u;
        const float2 f = tb[h & (kT - 1u)];
        const float w = (b2 ? rx : 1.f - rx) * (b1 ? ry : 1.f - ry) *
                        (b0 ? rz : 1.f - rz);
        a0 += w * f.x;
        a1 += w * f.y;
      }
      scr[laneLo * 32 + lv * 2 + 0] = (_Float16)a0;
      scr[laneLo * 32 + lv * 2 + 1] = (_Float16)a1;
    }
#endif
    LDS_FENCE();

    // ---- MLP-sigma layer 1: h = relu(enc(16x32) @ w1s(32x64)) ----
    const v16h aE = load_a(scr, 32, 0);
    v8f hAcc[4];
#pragma unroll
    for (int nt = 0; nt < 4; ++nt) {
      v8f z = {};
      hAcc[nt] = wmma_f16(aE, B1s[nt], z);
    }
    LDS_FENCE();
#pragma unroll
    for (int nt = 0; nt < 4; ++nt) store_c(scr, 64, nt * 16, hAcc[nt], true);
    LDS_FENCE();

    // ---- MLP-sigma layer 2: hf = h(16x64) @ w2s(64x16), K-chained WMMA ----
    {
      const v16h aH0 = load_a(scr, 64, 0);
      const v16h aH1 = load_a(scr, 64, 32);
      v8f hf = {};
      hf = wmma_f16(aH0, B2s[0], hf);
      hf = wmma_f16(aH1, B2s[1], hf);

      // alphas from column 0 (owned by lanes 0 and 16)
      if (laneLo == 0) {
#pragma unroll
        for (int r = 0; r < 8; ++r) {
          const float a = hf[r];
          out_alpha[s0 + r + hi * 8] = 1.f - __expf(-__expf(a) * kStep);
        }
      }
      LDS_FENCE();
      // features tile (16x32): cols 0..15 = hash_features
      store_c(scr, 32, 0, hf, false);
    }

    // ---- SH encode -> feature cols 16..31 (lanes 0-15, one sample each) ----
    if (hi == 0) {
      const float x = dir[(size_t)(s0 + laneLo) * 3 + 0];
      const float y = dir[(size_t)(s0 + laneLo) * 3 + 1];
      const float z = dir[(size_t)(s0 + laneLo) * 3 + 2];
      const float x2 = x * x, y2 = y * y, z2 = z * z;
      const float xy = x * y, yz = y * z, xz = x * z;
      float sh[16];
      sh[0]  = 0.28209479177387814f;
      sh[1]  = -0.48860251190291987f * y;
      sh[2]  = 0.48860251190291987f * z;
      sh[3]  = -0.48860251190291987f * x;
      sh[4]  = 1.0925484305920792f * xy;
      sh[5]  = -1.0925484305920792f * yz;
      sh[6]  = 0.94617469575756f * z2 - 0.31539156525252f;
      sh[7]  = -1.0925484305920792f * xz;
      sh[8]  = 0.5462742152960396f * (x2 - y2);
      sh[9]  = 0.5900435899266435f * y * (-3.f * x2 + y2);
      sh[10] = 2.890611442640554f * xy * z;
      sh[11] = 0.4570457994644657f * y * (1.f - 5.f * z2);
      sh[12] = 0.3731763325901154f * z * (5.f * z2 - 3.f);
      sh[13] = 0.4570457994644657f * x * (1.f - 5.f * z2);
      sh[14] = 1.445305721320277f * z * (x2 - y2);
      sh[15] = 0.5900435899266435f * x * (-x2 + 3.f * y2);
#pragma unroll
      for (int j = 0; j < 16; ++j)
        scr[laneLo * 32 + 16 + j] = (_Float16)sh[j];
    }
    LDS_FENCE();

    // ---- MLP-rgb layer 1: g1 = relu(features(16x32) @ w1r(32x64)) ----
    {
      const v16h aF = load_a(scr, 32, 0);
      v8f g1[4];
#pragma unroll
      for (int nt = 0; nt < 4; ++nt) {
        v8f z = {};
        g1[nt] = wmma_f16(aF, B1r[nt], z);
      }
      LDS_FENCE();
#pragma unroll
      for (int nt = 0; nt < 4; ++nt) store_c(scr, 64, nt * 16, g1[nt], true);
      LDS_FENCE();
    }

    // ---- MLP-rgb layer 2: g2 = relu(g1(16x64) @ w2r(64x64)) ----
    {
      const v16h aG0 = load_a(scr, 64, 0);
      const v16h aG1 = load_a(scr, 64, 32);
      v8f g2[4];
#pragma unroll
      for (int nt = 0; nt < 4; ++nt) {
        v8f z = {};
        z = wmma_f16(aG0, B2r[0][nt], z);
        z = wmma_f16(aG1, B2r[1][nt], z);
        g2[nt] = z;
      }
      LDS_FENCE();
#pragma unroll
      for (int nt = 0; nt < 4; ++nt) store_c(scr, 64, nt * 16, g2[nt], true);
      LDS_FENCE();
    }

    // ---- MLP-rgb layer 3 + sigmoid: rgb = sigmoid((g2 @ w3r)[:, :3]) ----
    {
      const v16h aG0 = load_a(scr, 64, 0);
      const v16h aG1 = load_a(scr, 64, 32);
      v8f o = {};
      o = wmma_f16(aG0, B3r[0], o);
      o = wmma_f16(aG1, B3r[1], o);
      if (laneLo < 3) {
#pragma unroll
        for (int r = 0; r < 8; ++r) {
          const float v = o[r];
          out_rgb[(size_t)(s0 + r + hi * 8) * 3 + laneLo] =
              1.f / (1.f + __expf(-v));
        }
      }
    }
    LDS_FENCE();
  }
}

extern "C" void kernel_launch(void* const* d_in, const int* in_sizes, int n_in,
                              void* d_out, int out_size, void* d_ws,
                              size_t ws_size, hipStream_t stream) {
  (void)in_sizes; (void)n_in; (void)out_size; (void)d_ws; (void)ws_size;
  const float* pos   = (const float*)d_in[0];
  const float* dir   = (const float*)d_in[1];
  const float* table = (const float*)d_in[2];
  const float* w1s   = (const float*)d_in[3];
  const float* w2s   = (const float*)d_in[4];
  const float* w1r   = (const float*)d_in[5];
  const float* w2r   = (const float*)d_in[6];
  const float* w3r   = (const float*)d_in[7];
  float* rgb   = (float*)d_out;
  float* alpha = rgb + (size_t)kN * 3;   // tuple return: rgbs then alphas

  dim3 grid(1024), block(128);           // 4096 wave32s, 4 tiles per wave
  InstantNGP_fused_kernel<<<grid, block, 0, stream>>>(
      pos, dir, table, w1s, w2s, w1r, w2r, w3r, rgb, alpha);
}
